// _NLM_Layer_43353399886150
// MI455X (gfx1250) — matrix operand order repaired and checked by reading, not compiled, not run on hardware
//
#include <hip/hip_runtime.h>

typedef __attribute__((ext_vector_type(16))) _Float16 v16h;
typedef __attribute__((ext_vector_type(8)))  float    v8f;

#define NOBJ 1024
#define TILES_PER_WAVE 8   // 8 waves * 8 tiles * 16 rows = 1024 rows = one full i per block

// fast sigmoid: v_mul + v_exp + v_add + v_rcp (all co-executable with WMMA)
__device__ __forceinline__ float sigmoidf_(float x) {
  const float e = __builtin_amdgcn_exp2f(x * -1.442695040888963f);
  return __builtin_amdgcn_rcpf(1.0f + e);
}

// convert 8 contiguous f32 -> fragment halfs [off..off+7]
__device__ __forceinline__ void cvt8(const float* __restrict__ p, v16h& d, int off) {
  const float4 a = *(const float4*)(p);
  const float4 b = *(const float4*)(p + 4);
  d[off+0]=(_Float16)a.x; d[off+1]=(_Float16)a.y; d[off+2]=(_Float16)a.z; d[off+3]=(_Float16)a.w;
  d[off+4]=(_Float16)b.x; d[off+5]=(_Float16)b.y; d[off+6]=(_Float16)b.z; d[off+7]=(_Float16)b.w;
}

// ---------------- arity 0 (tiny): reduce t1 cols + 40->32->8 MLP ----------------
__global__ void __launch_bounds__(256)
nlm_arity0(const float* __restrict__ t0, const float* __restrict__ t1,
           const float* __restrict__ W0a, const float* __restrict__ b0a,
           const float* __restrict__ W0b, const float* __restrict__ b0b,
           float* __restrict__ out0) {
  __shared__ float smin[256], smax[256];
  __shared__ float x0[40];
  __shared__ float hh[32];
  const int tid = threadIdx.x;
  const int p = tid & 15, g = tid >> 4;
  float vmin = 3.4e38f, vmax = -3.4e38f;
  for (int r = g; r < NOBJ; r += 16) {
    float v = t1[r*16 + p];
    vmin = fminf(vmin, v); vmax = fmaxf(vmax, v);
  }
  smin[tid]=vmin; smax[tid]=vmax;
  __syncthreads();
  for (int s = 128; s >= 16; s >>= 1) {
    if (tid < s) { smin[tid]=fminf(smin[tid],smin[tid+s]); smax[tid]=fmaxf(smax[tid],smax[tid+s]); }
    __syncthreads();
  }
  if (tid < 8)  x0[tid] = t0[tid];
  if (tid < 16) { x0[8+tid] = smin[tid]; x0[24+tid] = smax[tid]; }
  __syncthreads();
  if (tid < 32) {
    float acc = b0a[tid];
    for (int k = 0; k < 40; ++k) acc += x0[k] * W0a[k*32 + tid];
    hh[tid] = sigmoidf_(acc);
  }
  __syncthreads();
  if (tid < 8) {
    float acc = b0b[tid];
    for (int k = 0; k < 32; ++k) acc += hh[k] * W0b[k*8 + tid];
    out0[8+tid] = sigmoidf_(acc);
    out0[tid]   = t0[tid];
  }
}

// ---------------- arity 1 (small): per-object 56->32->16 MLP + residual ----------------
__global__ void __launch_bounds__(256)
nlm_arity1(const float* __restrict__ t0, const float* __restrict__ t1,
           const float* __restrict__ rmin, const float* __restrict__ rmax,
           const float* __restrict__ W1a, const float* __restrict__ b1a,
           const float* __restrict__ W1b, const float* __restrict__ b1b,
           float* __restrict__ out1) {
  const int j = blockIdx.x * blockDim.x + threadIdx.x;
  if (j >= NOBJ) return;
  float x[56];
  for (int k = 0; k < 8;  ++k) x[k]     = t0[k];
  for (int k = 0; k < 16; ++k) x[8+k]   = t1[j*16 + k];
  for (int k = 0; k < 16; ++k) x[24+k]  = rmin[j*16 + k];
  for (int k = 0; k < 16; ++k) x[40+k]  = rmax[j*16 + k];
  float h[32];
  for (int m = 0; m < 32; ++m) {
    float acc = b1a[m];
    for (int k = 0; k < 56; ++k) acc += x[k] * W1a[k*32 + m];
    h[m] = sigmoidf_(acc);
  }
  float* orow = out1 + j*32;
  for (int k = 0; k < 16; ++k) orow[k] = x[8+k];
  for (int o = 0; o < 16; ++o) {
    float acc = b1b[o];
    for (int k = 0; k < 32; ++k) acc += h[k] * W1b[k*16 + o];
    orow[16+o] = sigmoidf_(acc);
  }
}

// ---------------- arity 2 (dominant): WMMA MLP over N*N rows + fused t2 min/max ----------------
// Block bi covers all 1024 rows of i = bi (64 tiles). Row r = i*N + j,
// features: [ t1[i] | t2[i,j] | t1[j] | t2[j,i] ] (64).
// GEMM1: 4x v_wmma_f32_16x16x32_f16; GEMM2: 1x WMMA; residual-concat writeback.
// The residual loads stream every element of t2[i][*][*] through registers,
// so the axis=-2 min/max reduction is fused here (per-lane running -> LDS tree).
__global__ void __launch_bounds__(256)
nlm_arity2_wmma(const float* __restrict__ t1, const float* __restrict__ t2,
                const float* __restrict__ W2a, const float* __restrict__ b2a,
                const float* __restrict__ W2b, const float* __restrict__ b2b,
                float* __restrict__ out2,
                float* __restrict__ rmin, float* __restrict__ rmax) {
  __shared__ float hid[8][16][33];   // padded: avoids 16-stride bank conflicts
  __shared__ float ost[8][16][20];   // padded to 4-float multiple for aligned reads

  const int tid  = threadIdx.x;
  const int wave = tid >> 5;
  const int lane = tid & 31;
  const int hlf  = lane >> 4;        // lane half (K-sub-tile selector)
  const int nn   = lane & 15;        // B/C column, A row
  const int b8   = hlf * 8;

  // B fragments (32x16 f16): elem e <-> K = 8*hlf + e (e<8), 16 + 8*hlf + (e-8) (e>=8), column nn
  v16h bA00, bA01, bA10, bA11, bB;
  for (int e = 0; e < 16; ++e) {
    const int k = (e < 8) ? (8*hlf + e) : (16 + 8*hlf + (e - 8));
    bA00[e] = (_Float16)W2a[(k     )*32 + nn     ];
    bA01[e] = (_Float16)W2a[(k     )*32 + 16 + nn];
    bA10[e] = (_Float16)W2a[(k + 32)*32 + nn     ];
    bA11[e] = (_Float16)W2a[(k + 32)*32 + 16 + nn];
    bB  [e] = (_Float16)W2b[(k     )*16 + nn     ];
  }
  const float bias_a0 = b2a[nn];
  const float bias_a1 = b2a[16 + nn];
  const float bias_b  = b2b[nn];

  const int i = blockIdx.x;                 // one full object-i row per block

  // running min/max of t2[i][*][cols q..q+7] seen by this lane
  float lmin[8], lmax[8];
  for (int f = 0; f < 8; ++f) { lmin[f] = 3.4e38f; lmax[f] = -3.4e38f; }

  for (int tt = 0; tt < TILES_PER_WAVE; ++tt) {
    const int tile = (blockIdx.x * 8 + wave) * TILES_PER_WAVE + tt;
    const int r0 = tile << 4;               // first row of this 16-row tile
    const int jb = r0 & (NOBJ - 1);
    const int j  = jb + nn;                 // this lane owns A-row m = nn

    // A fragments (16x32 f16): lane half selects K-groups of 8
    v16h a0, a1;
    cvt8(t1 + i*16 + b8,                         a0, 0);  // feat 0..15  : t1[i]
    cvt8(t2 + ((size_t)i*NOBJ + j)*16 + b8,      a0, 8);  // feat 16..31 : t2[i][j]
    cvt8(t1 + j*16 + b8,                         a1, 0);  // feat 32..47 : t1[j]
    cvt8(t2 + ((size_t)j*NOBJ + i)*16 + b8,      a1, 8);  // feat 48..63 : t2[j][i]

    v8f c0, c1;
    for (int e = 0; e < 8; ++e) { c0[e] = bias_a0; c1[e] = bias_a1; }
    c0 = __builtin_amdgcn_wmma_f32_16x16x32_f16(false, a0, false, bA00, (short)0, c0, false, false);
    c0 = __builtin_amdgcn_wmma_f32_16x16x32_f16(false, a1, false, bA10, (short)0, c0, false, false);
    c1 = __builtin_amdgcn_wmma_f32_16x16x32_f16(false, a0, false, bA01, (short)0, c1, false, false);
    c1 = __builtin_amdgcn_wmma_f32_16x16x32_f16(false, a1, false, bA11, (short)0, c1, false, false);

    // hidden (C layout: row = 8*hlf+v, col = nn / 16+nn) -> LDS, with sigmoid
    for (int v = 0; v < 8; ++v) {
      hid[wave][8*hlf + v][nn]      = sigmoidf_(c0[v]);
      hid[wave][8*hlf + v][16 + nn] = sigmoidf_(c1[v]);
    }
    __syncthreads();

    // rebuild A fragment for GEMM2 from LDS (row m = nn)
    v16h a2;
    for (int e = 0; e < 8; ++e) {
      a2[e]     = (_Float16)hid[wave][nn][b8 + e];
      a2[8 + e] = (_Float16)hid[wave][nn][16 + b8 + e];
    }
    v8f c2;
    for (int e = 0; e < 8; ++e) c2[e] = bias_b;
    c2 = __builtin_amdgcn_wmma_f32_16x16x32_f16(false, a2, false, bB, (short)0, c2, false, false);

    for (int v = 0; v < 8; ++v)
      ost[wave][8*hlf + v][nn] = sigmoidf_(c2[v]);
    __syncthreads();

    // coalesced writeback: 2 lanes per output row, 8 floats each half
    const int m2 = lane >> 1;
    const int q  = (lane & 1) * 8;
    const size_t r = (size_t)r0 + m2;       // = i*N + jb + m2
    const float* src = t2 + r*16 + q;       // residual copy of t2[i][jb+m2]
    float* dst = out2 + r*32;
    const float4 s0 = *(const float4*)(src);
    const float4 s1 = *(const float4*)(src + 4);
    *(float4*)(dst + q)      = s0;
    *(float4*)(dst + q + 4)  = s1;
    // fused axis=-2 reduction on in-register residual data
    lmin[0] = fminf(lmin[0], s0.x); lmax[0] = fmaxf(lmax[0], s0.x);
    lmin[1] = fminf(lmin[1], s0.y); lmax[1] = fmaxf(lmax[1], s0.y);
    lmin[2] = fminf(lmin[2], s0.z); lmax[2] = fmaxf(lmax[2], s0.z);
    lmin[3] = fminf(lmin[3], s0.w); lmax[3] = fmaxf(lmax[3], s0.w);
    lmin[4] = fminf(lmin[4], s1.x); lmax[4] = fmaxf(lmax[4], s1.x);
    lmin[5] = fminf(lmin[5], s1.y); lmax[5] = fmaxf(lmax[5], s1.y);
    lmin[6] = fminf(lmin[6], s1.z); lmax[6] = fmaxf(lmax[6], s1.z);
    lmin[7] = fminf(lmin[7], s1.w); lmax[7] = fmaxf(lmax[7], s1.w);
    const float* hv = &ost[wave][m2][q];
    const float4 o0 = { hv[0], hv[1], hv[2], hv[3] };
    const float4 o1 = { hv[4], hv[5], hv[6], hv[7] };
    *(float4*)(dst + 16 + q)     = o0;
    *(float4*)(dst + 16 + q + 4) = o1;
  }

  // ---- block-wide min/max reduce (reuse LDS; hid/ost are dead now) ----
  __syncthreads();
  float* redmin = (float*)hid;   // needs 256*8 = 2048 floats (hid has 4224)
  float* redmax = (float*)ost;   // needs 2048 floats (ost has 2560)
  for (int f = 0; f < 8; ++f) { redmin[tid*8 + f] = lmin[f]; redmax[tid*8 + f] = lmax[f]; }
  __syncthreads();
  // element f of thread tid holds column (tid&1)*8 + f; reduce over g = tid>>1
  for (int s = 64; s >= 1; s >>= 1) {
    if ((tid >> 1) < s) {
      const int base = tid * 8, other = (tid + 2*s) * 8;
      for (int f = 0; f < 8; ++f) {
        redmin[base+f] = fminf(redmin[base+f], redmin[other+f]);
        redmax[base+f] = fmaxf(redmax[base+f], redmax[other+f]);
      }
    }
    __syncthreads();
  }
  if (tid < 16) {                 // tid0 holds cols 0..7 at [0..7], tid1 cols 8..15 at [8..15]
    rmin[i*16 + tid] = redmin[tid];
    rmax[i*16 + tid] = redmax[tid];
  }
}

extern "C" void kernel_launch(void* const* d_in, const int* in_sizes, int n_in,
                              void* d_out, int out_size, void* d_ws, size_t ws_size,
                              hipStream_t stream) {
  const float* t0  = (const float*)d_in[0];
  const float* t1  = (const float*)d_in[1];
  const float* t2  = (const float*)d_in[2];
  const float* W0a = (const float*)d_in[3];
  const float* b0a = (const float*)d_in[4];
  const float* W0b = (const float*)d_in[5];
  const float* b0b = (const float*)d_in[6];
  const float* W1a = (const float*)d_in[7];
  const float* b1a = (const float*)d_in[8];
  const float* W1b = (const float*)d_in[9];
  const float* b1b = (const float*)d_in[10];
  const float* W2a = (const float*)d_in[11];
  const float* b2a = (const float*)d_in[12];
  const float* W2b = (const float*)d_in[13];
  const float* b2b = (const float*)d_in[14];

  float* out0 = (float*)d_out;
  float* out1 = out0 + 16;
  float* out2 = out1 + (size_t)NOBJ * 32;

  float* rmin = (float*)d_ws;              // [N,16]
  float* rmax = rmin + NOBJ * 16;          // [N,16]  (128 KB scratch total)

  nlm_arity0<<<1, 256, 0, stream>>>(t0, t1, W0a, b0a, W0b, b0b, out0);
  // one block per object-i: computes out2 rows AND rmin/rmax[i] (fused reduction)
  nlm_arity2_wmma<<<NOBJ, 256, 0, stream>>>(t1, t2, W2a, b2a, W2b, b2b, out2, rmin, rmax);
  nlm_arity1<<<NOBJ / 256, 256, 0, stream>>>(t0, t1, rmin, rmax, W1a, b1a, W1b, b1b, out1);
}
